// MessagePassingNet_6356551598779
// MI455X (gfx1250) — matrix operand order repaired and checked by reading, not compiled, not run on hardware
//
#include <hip/hip_runtime.h>
#include <math.h>
#include <stdint.h>

// ---------------------------------------------------------------------------
// MI455X / gfx1250 implementation of the MPNN reference.
// Heavy matrix work (edge-network NNConv messages, conv_root, GRU, LSTM, fc1)
// runs on v_wmma_f32_16x16x32_bf16.  The per-edge message einsum is fused:
//   msg_e = (h_e (x) out[src_e]) . W2'   with W2' pre-transposed to bf16 [128][16512]
// so the 838MB `we` tensor is never materialized; B streams from L2 (4MB).
// ---------------------------------------------------------------------------

typedef __attribute__((ext_vector_type(16))) __bf16 bf16x16;
typedef __attribute__((ext_vector_type(8)))  __bf16 bf16x8;
typedef __attribute__((ext_vector_type(8)))  float  f32x8;

#define NNODES 3200
#define NEDGES 12800
#define NGRAPH 128
#define GS     128
#define KW     16512   // 16384 (W2 cols, k*128+i order) + 128 (en_b2 slab)

__device__ __forceinline__ float sigm(float x) { return 1.0f / (1.0f + __expf(-x)); }

// --- WMMA fragment loaders -------------------------------------------------
// A (16x32 bf16) from row-major [16][rowstride] buffer:
//   lane<16: M=lane, K = {0..7, 16..23};  lane>=16: M=lane-16, K = {8..15, 24..31}
__device__ __forceinline__ bf16x16 frag_a(const __bf16* base, int rowstride) {
  int l = threadIdx.x & 31;
  const __bf16* p = base + (l & 15) * rowstride + ((l >> 4) << 3);
  bf16x8 lo = *(const bf16x8*)p;
  bf16x8 hi = *(const bf16x8*)(p + 16);
  return __builtin_shufflevector(lo, hi, 0,1,2,3,4,5,6,7,8,9,10,11,12,13,14,15);
}
// B (32x16 bf16) from TRANSPOSED row-major [16 n][rowstride k] buffer:
//   lane<16: N=lane, K = {0..7, 16..23};  lane>=16: N=lane-16, K = {8..15, 24..31}
__device__ __forceinline__ bf16x16 frag_bT(const __bf16* base, int rowstride) {
  int l = threadIdx.x & 31;
  const __bf16* p = base + (l & 15) * rowstride + ((l >> 4) << 3);
  bf16x8 lo = *(const bf16x8*)p;
  bf16x8 hi = *(const bf16x8*)(p + 16);
  return __builtin_shufflevector(lo, hi, 0,1,2,3,4,5,6,7,8,9,10,11,12,13,14,15);
}

// --- generic 16x16-tile WMMA GEMM: C[M,N] = op(A[M,K] @ B[K,N] + bias) ------
// one wave per block; K multiple of 32, M,N multiples of 16 (always true here)
template<int RELU, int ACCUM>
__global__ __launch_bounds__(32) void gemm16(const float* __restrict__ A,
                                             const float* __restrict__ Bm,
                                             const float* __restrict__ bias,
                                             float* __restrict__ C,
                                             int M, int K, int N) {
  __shared__ __align__(16) __bf16 Asm[16][40];
  __shared__ __align__(16) __bf16 Bsm[16][40];
  const int tid = threadIdx.x;
  const int m0 = blockIdx.x * 16, n0 = blockIdx.y * 16;
  f32x8 acc = {};
  for (int k0 = 0; k0 < K; k0 += 32) {
    for (int idx = tid; idx < 512; idx += 32) {       // stage A tile -> bf16
      int r = idx >> 5, c = idx & 31;
      Asm[r][c] = (__bf16)A[(size_t)(m0 + r) * K + k0 + c];
    }
    for (int idx = tid; idx < 512; idx += 32) {       // stage B tile transposed
      int n = idx >> 5, k = idx & 31;
      Bsm[n][k] = (__bf16)Bm[(size_t)(k0 + k) * N + n0 + n];
    }
    __syncthreads();
    bf16x16 a = frag_a(&Asm[0][0], 40);
    bf16x16 b = frag_bT(&Bsm[0][0], 40);
    acc = __builtin_amdgcn_wmma_f32_16x16x32_bf16(false, a, false, b,
                                                  (short)0, acc, false, false);
    __syncthreads();
  }
  const int n  = n0 + (tid & 15);
  const int mb = (tid >> 4) << 3;
  const float bv = bias ? bias[n] : 0.0f;
#pragma unroll
  for (int j = 0; j < 8; ++j) {
    size_t off = (size_t)(m0 + mb + j) * N + n;
    float v = acc[j] + bv;
    if (ACCUM) v += C[off];
    if (RELU)  v = fmaxf(v, 0.0f);
    C[off] = v;
  }
}

// --- fused NNConv message kernel -------------------------------------------
// block = 8 waves, one 16-edge tile.  wave w owns output columns [16w,16w+16).
// A is generated on the fly: P[m][kk] = h[m][k] * out_src[m][ibase+kk].
// The per-lane out_src values (4 subchunks x 16) are loop-invariant -> hoisted
// into 64 VGPRs.  B is software-pipelined one 32-K chunk ahead so the
// s_wait_loadcnt lands a full iteration away from the consuming WMMA.
// en_b2 handled by a branch-free 4-chunk epilogue (h == 1).
__global__ __launch_bounds__(256) void msg_kernel(const float* __restrict__ out,
                                                  const float* __restrict__ he,
                                                  const __bf16* __restrict__ W2t,
                                                  const int* __restrict__ src,
                                                  const int* __restrict__ dst,
                                                  float* __restrict__ agg) {
  __shared__ __align__(16) float Os[16][GS];
  __shared__ __align__(16) float Hs[16][GS];
  __shared__ int dsts[16];
  const int tid = threadIdx.x;
  const int e0 = blockIdx.x * 16;
  for (int idx = tid; idx < 16 * GS; idx += 256) {
    int m = idx >> 7, c = idx & 127;
    Os[m][c] = out[(size_t)src[e0 + m] * GS + c];
    Hs[m][c] = he[(size_t)(e0 + m) * GS + c];
  }
  if (tid < 16) dsts[tid] = dst[e0 + tid];
  __syncthreads();

  const int wave    = tid >> 5;        // n-tile 0..7
  const int l       = tid & 31;
  const int row     = l & 15;          // A: M row / B: N column
  const int hi_half = l >> 4;
  const __bf16* brow = W2t + (size_t)(wave * 16 + row) * KW + (hi_half << 4);
  const float*  orow = Os[row];

  // hoist the 4 x 16 loop-invariant out_src values for this lane into VGPRs
  float oreg[4][16];
#pragma unroll
  for (int sub = 0; sub < 4; ++sub) {
    const float* p = orow + (sub << 5) + (hi_half << 4);
#pragma unroll
    for (int j = 0; j < 8; ++j) { oreg[sub][j] = p[j]; oreg[sub][8 + j] = p[8 + j]; }
  }

  f32x8 acc = {};
  const __bf16* q = brow;
  // prime the B pipeline with chunk 0
  bf16x8 blo = *(const bf16x8*)q;
  bf16x8 bhi = *(const bf16x8*)(q + 8);
  for (int k = 0; k < 128; ++k) {
    const float hk = Hs[row][k];
#pragma unroll
    for (int sub = 0; sub < 4; ++sub) {
      bf16x16 a;
#pragma unroll
      for (int j = 0; j < 16; ++j) a[j] = (__bf16)(hk * oreg[sub][j]);
      bf16x16 b = __builtin_shufflevector(blo, bhi,
                                          0,1,2,3,4,5,6,7,8,9,10,11,12,13,14,15);
      // prefetch next chunk (at k==127,sub==3 this is the bias slab, in-bounds)
      const __bf16* qn = q + ((sub + 1) << 5);
      blo = *(const bf16x8*)qn;
      bhi = *(const bf16x8*)(qn + 8);
      acc = __builtin_amdgcn_wmma_f32_16x16x32_bf16(false, a, false, b,
                                                    (short)0, acc, false, false);
    }
    q += 128;
  }
  // bias epilogue: 4 chunks against the en_b2 slab, h == 1 (q == brow + 16384)
#pragma unroll
  for (int sub = 0; sub < 4; ++sub) {
    bf16x16 a;
#pragma unroll
    for (int j = 0; j < 16; ++j) a[j] = (__bf16)oreg[sub][j];
    bf16x16 b = __builtin_shufflevector(blo, bhi,
                                        0,1,2,3,4,5,6,7,8,9,10,11,12,13,14,15);
    if (sub < 3) {                      // stay inside the KW allocation
      const __bf16* qn = q + ((sub + 1) << 5);
      blo = *(const bf16x8*)qn;
      bhi = *(const bf16x8*)(qn + 8);
    }
    acc = __builtin_amdgcn_wmma_f32_16x16x32_bf16(false, a, false, b,
                                                  (short)0, acc, false, false);
  }

  const int ncol = wave * 16 + (l & 15);
#pragma unroll
  for (int j = 0; j < 8; ++j) {
    int m = j + (hi_half << 3);
    atomicAdd(&agg[(size_t)dsts[m] * GS + ncol], acc[j]);
  }
}

// --- small VALU kernels -----------------------------------------------------
__global__ void lin0_kernel(const float* x, const float* w, const float* b, float* out) {
  int i = blockIdx.x * blockDim.x + threadIdx.x;
  if (i >= NNODES * GS) return;
  int n = i >> 7, c = i & 127;
  float s = b[c];
#pragma unroll
  for (int k = 0; k < 14; ++k) s += x[n * 14 + k] * w[k * GS + c];
  out[i] = fmaxf(s, 0.0f);
}

__global__ void edgeh_kernel(const float* ea, const float* w, const float* b, float* he) {
  int i = blockIdx.x * blockDim.x + threadIdx.x;
  if (i >= NEDGES * GS) return;
  int e = i >> 7, c = i & 127;
  float s = b[c];
#pragma unroll
  for (int k = 0; k < 4; ++k) s += ea[e * 4 + k] * w[k * GS + c];
  he[i] = fmaxf(s, 0.0f);
}

// W2t[o][k*128+i] = en_w2[k][i*128+o];  W2t[o][16384+i] = en_b2[i*128+o]
__global__ void pack_w2_kernel(const float* w2, const float* b2, __bf16* W2t) {
  int i = blockIdx.x * blockDim.x + threadIdx.x;
  if (i >= GS * KW) return;
  int o = i / KW, col = i % KW;
  float v;
  if (col < 16384) { int k = col >> 7, ii = col & 127; v = w2[(size_t)k * 16384 + ii * GS + o]; }
  else             { int ii = col - 16384;             v = b2[ii * GS + o]; }
  W2t[i] = (__bf16)v;
}

__global__ void deg_kernel(const int* dst, float* deg) {
  int e = blockIdx.x * blockDim.x + threadIdx.x;
  if (e < NEDGES) atomicAdd(&deg[dst[e]], 1.0f);
}
__global__ void invdeg_kernel(float* deg) {
  int n = blockIdx.x * blockDim.x + threadIdx.x;
  if (n < NNODES) deg[n] = 1.0f / fmaxf(deg[n], 1.0f);
}

__global__ void combine_kernel(const float* agg, const float* rootb, const float* cbias,
                               const float* invdeg, float* m) {
  int i = blockIdx.x * blockDim.x + threadIdx.x;
  if (i >= NNODES * GS) return;
  int n = i >> 7, c = i & 127;
  m[i] = fmaxf(agg[i] * invdeg[n] + rootb[i] + cbias[c], 0.0f);
}

__global__ void gru_kernel(const float* gi, const float* gh, float* h) {
  int i = blockIdx.x * blockDim.x + threadIdx.x;
  if (i >= NNODES * GS) return;
  int n = i >> 7, c = i & 127;
  const float* a = gi + (size_t)n * 384;
  const float* b = gh + (size_t)n * 384;
  float r  = sigm(a[c] + b[c]);
  float z  = sigm(a[128 + c] + b[128 + c]);
  float nn = tanhf(a[256 + c] + r * b[256 + c]);
  h[i] = (1.0f - z) * nn + z * h[i];
}

__global__ void lstm_kernel(const float* gates, float* qc, float* qh) {
  int i = blockIdx.x * blockDim.x + threadIdx.x;
  if (i >= NGRAPH * GS) return;
  int b = i >> 7, c = i & 127;
  const float* g = gates + (size_t)b * 512;
  float ig = sigm(g[c]), fg = sigm(g[128 + c]);
  float gg = tanhf(g[256 + c]), og = sigm(g[384 + c]);
  float cc = fg * qc[i] + ig * gg;
  qc[i] = cc;
  qh[i] = og * tanhf(cc);
}

__global__ void escore_kernel(const float* out, const float* qh, const int* batch, float* e) {
  int n = blockIdx.x * blockDim.x + threadIdx.x;
  if (n >= NNODES) return;
  const float* o = out + (size_t)n * GS;
  const float* q = qh + (size_t)batch[n] * GS;
  float s = 0.0f;
  for (int c = 0; c < GS; ++c) s += o[c] * q[c];
  e[n] = s;
}

__global__ void segmax_kernel(const float* e, const int* batch, float* emax) {
  __shared__ float red[128];
  int b = blockIdx.x;
  float m = -INFINITY;
  for (int n = threadIdx.x; n < NNODES; n += 128)
    if (batch[n] == b) m = fmaxf(m, e[n]);
  red[threadIdx.x] = m; __syncthreads();
  for (int s = 64; s > 0; s >>= 1) {
    if (threadIdx.x < s) red[threadIdx.x] = fmaxf(red[threadIdx.x], red[threadIdx.x + s]);
    __syncthreads();
  }
  if (threadIdx.x == 0) emax[b] = isfinite(red[0]) ? red[0] : 0.0f;
}

__global__ void aun_kernel(const float* e, const float* emax, const int* batch, float* aun) {
  int n = blockIdx.x * blockDim.x + threadIdx.x;
  if (n < NNODES) aun[n] = __expf(e[n] - emax[batch[n]]);
}

__global__ void segsum_kernel(const float* aun, const int* batch, float* denom) {
  __shared__ float red[128];
  int b = blockIdx.x;
  float s = 0.0f;
  for (int n = threadIdx.x; n < NNODES; n += 128)
    if (batch[n] == b) s += aun[n];
  red[threadIdx.x] = s; __syncthreads();
  for (int st = 64; st > 0; st >>= 1) {
    if (threadIdx.x < st) red[threadIdx.x] += red[threadIdx.x + st];
    __syncthreads();
  }
  if (threadIdx.x == 0) denom[b] = red[0];
}

__global__ void rread_kernel(const float* aun, const float* denom, const int* batch,
                             const float* out, float* rread) {
  int i = blockIdx.x * blockDim.x + threadIdx.x;
  if (i >= NNODES * GS) return;
  int n = i >> 7, c = i & 127;
  int b = batch[n];
  atomicAdd(&rread[(size_t)b * GS + c], (aun[n] / denom[b]) * out[i]);
}

__global__ void qstar_kernel(const float* qh, const float* rread, float* qstar) {
  int i = blockIdx.x * blockDim.x + threadIdx.x;
  if (i >= NGRAPH * 256) return;
  int b = i >> 8, c = i & 255;
  qstar[i] = (c < 128) ? qh[b * GS + c] : rread[b * GS + (c - 128)];
}

__global__ void fc2_kernel(const float* fc1o, const float* w, const float* b, float* y) {
  int g = blockIdx.x * blockDim.x + threadIdx.x;
  if (g >= NGRAPH) return;
  float s = b[0];
  for (int c = 0; c < 128; ++c) s += fc1o[g * 128 + c] * w[c];
  y[g] = s;
}

// ---------------------------------------------------------------------------
static inline char* bump(char*& p, size_t bytes) {
  char* r = (char*)(((uintptr_t)p + 255) & ~(uintptr_t)255);
  p = r + bytes;
  return r;
}
static inline int cdiv(int a, int b) { return (a + b - 1) / b; }

extern "C" void kernel_launch(void* const* d_in, const int* in_sizes, int n_in,
                              void* d_out, int out_size, void* d_ws, size_t ws_size,
                              hipStream_t stream) {
  const float* x        = (const float*)d_in[0];
  const float* ea       = (const float*)d_in[1];
  const float* lin0_w   = (const float*)d_in[2];
  const float* lin0_b   = (const float*)d_in[3];
  const float* en_w1    = (const float*)d_in[4];
  const float* en_b1    = (const float*)d_in[5];
  const float* en_w2    = (const float*)d_in[6];
  const float* en_b2    = (const float*)d_in[7];
  const float* conv_root= (const float*)d_in[8];
  const float* conv_bias= (const float*)d_in[9];
  const float* gru_w_ih = (const float*)d_in[10];
  const float* gru_w_hh = (const float*)d_in[11];
  const float* gru_b_ih = (const float*)d_in[12];
  const float* gru_b_hh = (const float*)d_in[13];
  const float* lstm_w_ih= (const float*)d_in[14];
  const float* lstm_w_hh= (const float*)d_in[15];
  const float* lstm_b_ih= (const float*)d_in[16];
  const float* lstm_b_hh= (const float*)d_in[17];
  const float* fc1_w    = (const float*)d_in[18];
  const float* fc1_b    = (const float*)d_in[19];
  const float* fc2_w    = (const float*)d_in[20];
  const float* fc2_b    = (const float*)d_in[21];
  const int*   eindex   = (const int*)d_in[22];
  const int*   batch    = (const int*)d_in[23];
  const int* src = eindex;
  const int* dst = eindex + NEDGES;
  float* y = (float*)d_out;

  char* w = (char*)d_ws;
  float*  out    = (float*) bump(w, (size_t)NNODES * GS * 4);
  float*  he     = (float*) bump(w, (size_t)NEDGES * GS * 4);
  __bf16* W2t    = (__bf16*)bump(w, (size_t)GS * KW * 2);
  float*  agg    = (float*) bump(w, (size_t)NNODES * GS * 4);
  float*  rootb  = (float*) bump(w, (size_t)NNODES * GS * 4);
  float*  mbuf   = (float*) bump(w, (size_t)NNODES * GS * 4);
  float*  gi     = (float*) bump(w, (size_t)NNODES * 384 * 4);
  float*  gh     = (float*) bump(w, (size_t)NNODES * 384 * 4);
  float*  invdeg = (float*) bump(w, (size_t)NNODES * 4);
  float*  qh     = (float*) bump(w, (size_t)NGRAPH * GS * 4);
  float*  qc     = (float*) bump(w, (size_t)NGRAPH * GS * 4);
  float*  qstar  = (float*) bump(w, (size_t)NGRAPH * 256 * 4);
  float*  gates  = (float*) bump(w, (size_t)NGRAPH * 512 * 4);
  float*  e      = (float*) bump(w, (size_t)NNODES * 4);
  float*  aun    = (float*) bump(w, (size_t)NNODES * 4);
  float*  emax   = (float*) bump(w, (size_t)NGRAPH * 4);
  float*  denom  = (float*) bump(w, (size_t)NGRAPH * 4);
  float*  rread  = (float*) bump(w, (size_t)NGRAPH * GS * 4);
  float*  fc1o   = (float*) bump(w, (size_t)NGRAPH * GS * 4);

  const int NE = NNODES * GS;   // 409600
  const int EE = NEDGES * GS;   // 1638400

  // ---- prologue: projections, edge-net packing, degrees -------------------
  lin0_kernel<<<cdiv(NE, 256), 256, 0, stream>>>(x, lin0_w, lin0_b, out);
  edgeh_kernel<<<cdiv(EE, 256), 256, 0, stream>>>(ea, en_w1, en_b1, he);
  pack_w2_kernel<<<cdiv(GS * KW, 256), 256, 0, stream>>>(en_w2, en_b2, W2t);
  hipMemsetAsync(invdeg, 0, (size_t)NNODES * 4, stream);
  deg_kernel<<<cdiv(NEDGES, 256), 256, 0, stream>>>(dst, invdeg);
  invdeg_kernel<<<cdiv(NNODES, 256), 256, 0, stream>>>(invdeg);

  // ---- 3 message-passing iterations (out == h throughout) -----------------
  for (int it = 0; it < 3; ++it) {
    hipMemsetAsync(agg, 0, (size_t)NE * 4, stream);
    msg_kernel<<<NEDGES / 16, 256, 0, stream>>>(out, he, W2t, src, dst, agg);
    gemm16<0, 0><<<dim3(NNODES / 16, GS / 16), 32, 0, stream>>>(
        out, conv_root, nullptr, rootb, NNODES, GS, GS);
    combine_kernel<<<cdiv(NE, 256), 256, 0, stream>>>(agg, rootb, conv_bias, invdeg, mbuf);
    gemm16<0, 0><<<dim3(NNODES / 16, 384 / 16), 32, 0, stream>>>(
        mbuf, gru_w_ih, gru_b_ih, gi, NNODES, GS, 384);
    gemm16<0, 0><<<dim3(NNODES / 16, 384 / 16), 32, 0, stream>>>(
        out, gru_w_hh, gru_b_hh, gh, NNODES, GS, 384);
    gru_kernel<<<cdiv(NE, 256), 256, 0, stream>>>(gi, gh, out);
  }

  // ---- Set2Set pooling ----------------------------------------------------
  hipMemsetAsync(qh, 0, (size_t)NGRAPH * GS * 4, stream);
  hipMemsetAsync(qc, 0, (size_t)NGRAPH * GS * 4, stream);
  hipMemsetAsync(qstar, 0, (size_t)NGRAPH * 256 * 4, stream);
  for (int t = 0; t < 3; ++t) {
    gemm16<0, 0><<<dim3(NGRAPH / 16, 512 / 16), 32, 0, stream>>>(
        qstar, lstm_w_ih, lstm_b_ih, gates, NGRAPH, 256, 512);
    gemm16<0, 1><<<dim3(NGRAPH / 16, 512 / 16), 32, 0, stream>>>(
        qh, lstm_w_hh, lstm_b_hh, gates, NGRAPH, GS, 512);
    lstm_kernel<<<cdiv(NGRAPH * GS, 256), 256, 0, stream>>>(gates, qc, qh);
    escore_kernel<<<cdiv(NNODES, 256), 256, 0, stream>>>(out, qh, batch, e);
    segmax_kernel<<<NGRAPH, 128, 0, stream>>>(e, batch, emax);
    aun_kernel<<<cdiv(NNODES, 256), 256, 0, stream>>>(e, emax, batch, aun);
    segsum_kernel<<<NGRAPH, 128, 0, stream>>>(aun, batch, denom);
    hipMemsetAsync(rread, 0, (size_t)NGRAPH * GS * 4, stream);
    rread_kernel<<<cdiv(NE, 256), 256, 0, stream>>>(aun, denom, batch, out, rread);
    qstar_kernel<<<cdiv(NGRAPH * 256, 256), 256, 0, stream>>>(qh, rread, qstar);
  }

  // ---- readout ------------------------------------------------------------
  gemm16<1, 0><<<dim3(NGRAPH / 16, GS / 16), 32, 0, stream>>>(
      qstar, fc1_w, fc1_b, fc1o, NGRAPH, 256, GS);
  fc2_kernel<<<1, NGRAPH, 0, stream>>>(fc1o, fc2_w, fc2_b, y);
}
